// RIMNetwork_specialized_multHead_68736656605806
// MI455X (gfx1250) — matrix operand order repaired and checked by reading, not compiled, not run on hardware
//
#include <hip/hip_runtime.h>
#include <hip/hip_bf16.h>
#include <math.h>

// ---------------- problem constants ----------------
#define B_    256
#define T_    128
#define I_    128
#define H_    256
#define M_    6
#define NH_   4
#define KD_   16     // attention key dim
#define R_    16     // low-rank dim
#define TASK_ 32
#define SENS_ 96
#define O_    64
#define BT    16     // batch tile per workgroup (WMMA M)

// packed bf16 weight offsets (ushort elements) inside d_ws.
// pack layout per matrix: [ntile][kchunk][lane(32)][16 bf16]  (fragment-native)
#define OFF_WT   0L        // W_in_task     2 * (32x256)   = 16384
#define OFF_WS   16384L    // W_in_sensory  2 * (96x256)   = 49152
#define OFF_WIP  65536L    // Wip           6 * (256x256)  = 393216
#define OFF_U    458752L   // U             6 * (256x16)   = 24576
#define OFF_VLR  483328L   // Vlr           6 * (32x256)   = 49152 (K padded 16->32)
#define OFF_WQ   532480L   // W_Q          24 * (256x16)   = 98304
#define OFF_WK   630784L   // W_K          24 * (256x16)   = 98304
#define OFF_WV   729088L   // W_V          24 * (256x256)  = 1572864
#define OFF_WO   2301952L  // Wout^T            (512x64)   = 32768

typedef __attribute__((ext_vector_type(16))) __bf16        v16bf;
typedef __attribute__((ext_vector_type(8)))  float         v8f;
typedef __attribute__((ext_vector_type(4)))  float         v4f;
typedef __attribute__((ext_vector_type(4)))  unsigned int  v4u;
typedef __attribute__((ext_vector_type(8)))  unsigned int  v8u;

// RNE f32->bf16 (used only in the one-time weight pack prepass)
__device__ __forceinline__ unsigned short f2bf_rne(float f) {
    unsigned int u = __builtin_bit_cast(unsigned int, f);
    unsigned int r = u + 0x7FFFu + ((u >> 16) & 1u);
    return (unsigned short)(r >> 16);
}
// Truncating f32->bf16 (activations; lowers to shift / d16_hi store)
__device__ __forceinline__ unsigned short f2bf_t(float f) {
    return (unsigned short)(__builtin_bit_cast(unsigned int, f) >> 16);
}
// Pack two f32 into two truncated bf16 with a single v_perm_b32
__device__ __forceinline__ unsigned int pack2bf(float lo, float hi) {
    return __builtin_amdgcn_perm(__builtin_bit_cast(unsigned int, hi),
                                 __builtin_bit_cast(unsigned int, lo), 0x07060302u);
}
__device__ __forceinline__ float bf2f(unsigned short h) {
    unsigned int u = ((unsigned int)h) << 16;
    return __builtin_bit_cast(float, u);
}
__device__ __forceinline__ v8f zero8() {
    v8f z;
#pragma unroll
    for (int i = 0; i < 8; ++i) z[i] = 0.0f;
    return z;
}

// ---- B fragment: one 32-byte contiguous load per lane from packed weights ----
__device__ __forceinline__ v16bf load_b_pk(const unsigned short* __restrict__ mat,
                                           int ntile, int kchunk, int nKchunks) {
    long off = (((long)ntile * nKchunks + kchunk) * 32 + (threadIdx.x & 31)) * 16;
    v8u raw = *(const v8u*)(mat + off);
    return __builtin_bit_cast(v16bf, raw);
}

// ---- A fragment (16x32 bf16) from LDS: two 16-byte loads per lane ----
// gfx1250 A layout: lanes 0-15 -> K kb..kb+7, kb+16..kb+23 with kb=0; lanes 16-31 kb=8
__device__ __forceinline__ v16bf load_a_lds(const unsigned short* p, int rowStride, int kofs) {
    int lane = threadIdx.x & 31;
    int row  = lane & 15;
    int kb   = (lane >> 4) * 8;
    const unsigned short* q = p + row * rowStride + kofs + kb;
    v4u lo = *(const v4u*)q;          // k = kb .. kb+7
    v4u hi = *(const v4u*)(q + 16);   // k = kb+16 .. kb+23
    v8u raw;
#pragma unroll
    for (int j = 0; j < 4; ++j) { raw[j] = lo[j]; raw[4 + j] = hi[j]; }
    return __builtin_bit_cast(v16bf, raw);
}

// ---- A fragment with K=16 (upper half zero): one 16-byte load ----
__device__ __forceinline__ v16bf load_a_lds_k16(const unsigned short* p, int rowStride) {
    int lane = threadIdx.x & 31;
    int row  = lane & 15;
    int kb   = (lane >> 4) * 8;
    v4u lo = *(const v4u*)(p + row * rowStride + kb);
    v8u raw;
#pragma unroll
    for (int j = 0; j < 4; ++j) { raw[j] = lo[j]; raw[4 + j] = 0u; }
    return __builtin_bit_cast(v16bf, raw);
}

// ---- A fragment from f32 global (x): four 16-byte loads + 8x v_perm pack ----
__device__ __forceinline__ v16bf load_a_xf32(const float* p, long rowStride, int kofs) {
    int lane = threadIdx.x & 31;
    int row  = lane & 15;
    int kb   = (lane >> 4) * 8;
    const float* q = p + (long)row * rowStride + kofs + kb;
    v4f f0 = *(const v4f*)q;          // frag elems 0..3
    v4f f1 = *(const v4f*)(q + 4);    // frag elems 4..7
    v4f f2 = *(const v4f*)(q + 16);   // frag elems 8..11
    v4f f3 = *(const v4f*)(q + 20);   // frag elems 12..15
    v8u raw;
    raw[0] = pack2bf(f0[0], f0[1]);
    raw[1] = pack2bf(f0[2], f0[3]);
    raw[2] = pack2bf(f1[0], f1[1]);
    raw[3] = pack2bf(f1[2], f1[3]);
    raw[4] = pack2bf(f2[0], f2[1]);
    raw[5] = pack2bf(f2[2], f2[3]);
    raw[6] = pack2bf(f3[0], f3[1]);
    raw[7] = pack2bf(f3[2], f3[3]);
    return __builtin_bit_cast(v16bf, raw);
}

__device__ __forceinline__ v8f wmma_bf16(v16bf a, v16bf b, v8f c) {
    return __builtin_amdgcn_wmma_f32_16x16x32_bf16(false, a, false, b, (short)0, c, false, false);
}

// -------- prepass: pack f32 weights into fragment-native bf16 layout --------
// dst[mat][ntile][kchunk][lane][e] = src[mat][k*strideK + n*strideN] (0 if k>=Ktot)
__global__ void rim_pack_b_kernel(const float* __restrict__ src, unsigned short* __restrict__ dst,
                                  int nMat, long srcMatStride, long dstMatStride,
                                  int Ktot, int Ntot, long strideK, long strideN, int Kpad) {
    long i = (long)blockIdx.x * blockDim.x + threadIdx.x;
    int  nK  = Kpad / 32;
    long per = (long)(Ntot / 16) * nK * 512;
    if (i >= per * nMat) return;
    int  mat = (int)(i / per);
    long rcd = i % per;
    int  e    = (int)(rcd & 15);
    int  lane = (int)((rcd >> 4) & 31);
    long chnk = rcd >> 9;                 // [ntile][kchunk]
    int  kchunk = (int)(chnk % nK);
    int  ntile  = (int)(chnk / nK);
    int  n  = ntile * 16 + (lane & 15);
    int  k  = kchunk * 32 + (lane >> 4) * 8 + (e & 7) + (e >> 3) * 16;
    float v = (k < Ktot) ? src[(long)mat * srcMatStride + (long)k * strideK + (long)n * strideN]
                         : 0.0f;
    dst[(long)mat * dstMatStride + rcd] = f2bf_rne(v);
}

// -------- persistent per-batch-tile recurrent kernel --------
__global__ __launch_bounds__(256) void rim_scan_kernel(
    const float* __restrict__ x,
    const float* __restrict__ bip,  const float* __restrict__ bias,
    const float* __restrict__ bout,
    const unsigned short* __restrict__ wbase,
    float* __restrict__ outp, float* __restrict__ hidp,
    float* __restrict__ ainp, float* __restrict__ acmp,
    float* __restrict__ awp) {

    __shared__ __align__(16) unsigned short sHst[BT][M_ * H_];  // carry H (bf16)   49152 B
    __shared__ __align__(16) unsigned short sHin[BT][M_ * H_];  // tanh out         49152 B
    __shared__ __align__(16) unsigned short sAin[BT][M_ * H_];  // A_in             49152 B
    __shared__ __align__(16) unsigned short sHcm[BT][2 * H_];   // H_comm mods 4,5  16384 B
    __shared__ __align__(16) unsigned short sTmp[M_][BT][R_];   // Hst@U            3072 B
    __shared__ __align__(16) float sQ [NH_][M_][BT][KD_];       //                  24576 B
    __shared__ __align__(16) float sKt[NH_][M_][BT][KD_];       //                  24576 B
    __shared__ __align__(16) float sAw[NH_][M_][M_][BT];        // blended attn      9216 B

    const int tid  = threadIdx.x;
    const int wave = tid >> 5;
    const int lane = tid & 31;
    const int col  = lane & 15;
    const int rb   = (lane >> 4) * 8;         // C-frag row base
    const int b0   = blockIdx.x * BT;

    for (int i = tid; i < BT * M_ * H_; i += 256) {
        (&sHst[0][0])[i] = 0;
        (&sAin[0][0])[i] = 0;
    }
    for (int i = tid; i < NH_ * M_ * M_ * BT; i += 256) (&sAw[0][0][0][0])[i] = 0.0f;
    __syncthreads();

    for (int t = 0; t < T_; ++t) {
        // ---- Phase A: A_in = [x_task@W_in_task | x_sens@W_in_sensory | 0] ----
        for (int task = wave; task < 64; task += 8) {
            int m = task >> 4, c = task & 15;
            v8f acc = zero8();
            if (m < 2) {
                const float* xa = x + (long)b0 * T_ * I_ + (long)t * I_ + SENS_;
                const unsigned short* wb = wbase + OFF_WT + (long)m * 32 * 256;
                v16bf a = load_a_xf32(xa, (long)T_ * I_, 0);
                v16bf b = load_b_pk(wb, c, 0, 1);
                acc = wmma_bf16(a, b, acc);
            } else {
                const float* xa = x + (long)b0 * T_ * I_ + (long)t * I_;
                const unsigned short* wb = wbase + OFF_WS + (long)(m - 2) * 96 * 256;
#pragma unroll
                for (int kc = 0; kc < 3; ++kc) {
                    v16bf a = load_a_xf32(xa, (long)T_ * I_, kc * 32);
                    v16bf b = load_b_pk(wb, c, kc, 3);
                    acc = wmma_bf16(a, b, acc);
                }
            }
#pragma unroll
            for (int r = 0; r < 8; ++r) {
                int row = rb + r;
                float v = acc[r];
                ainp[(((long)(b0 + row) * T_ + t) * M_ + m) * H_ + c * 16 + col] = v;
                sAin[row][m * H_ + c * 16 + col] = f2bf_t(v);
            }
        }
        for (int i = tid; i < BT * 2 * H_; i += 256) {       // zero modules 4,5
            int row = i / (2 * H_);
            int rem = i % (2 * H_);
            int m   = 4 + rem / H_;
            ainp[(((long)(b0 + row) * T_ + t) * M_ + m) * H_ + (rem % H_)] = 0.0f;
        }
        __syncthreads();

        // ---- Phase B1: tmp = Hst @ U[m]  (BT x 16) ----
        for (int m = wave; m < M_; m += 8) {
            v8f acc = zero8();
            const unsigned short* ub = wbase + OFF_U + (long)m * 256 * 16;
#pragma unroll
            for (int kc = 0; kc < 8; ++kc) {
                v16bf a = load_a_lds(&sHst[0][m * H_], M_ * H_, kc * 32);
                v16bf b = load_b_pk(ub, 0, kc, 8);
                acc = wmma_bf16(a, b, acc);
            }
#pragma unroll
            for (int r = 0; r < 8; ++r) sTmp[m][rb + r][col] = f2bf_t(acc[r]);
        }
        __syncthreads();

        // ---- Phase B2: H_in = tanh(A_in@Wip + tmp@Vlr + bip + bias) ----
        for (int task = wave; task < 96; task += 8) {
            int m = task / 16, c = task % 16;
            v8f acc = zero8();
            const unsigned short* wip = wbase + OFF_WIP + (long)m * 256 * 256;
#pragma unroll
            for (int kc = 0; kc < 8; ++kc) {
                v16bf a = load_a_lds(&sAin[0][m * H_], M_ * H_, kc * 32);
                v16bf b = load_b_pk(wip, c, kc, 8);
                acc = wmma_bf16(a, b, acc);
            }
            {   // low-rank K=16 (zero-padded in both operands)
                v16bf a = load_a_lds_k16(&sTmp[m][0][0], R_);
                v16bf b = load_b_pk(wbase + OFF_VLR + (long)m * 32 * 256, c, 0, 1);
                acc = wmma_bf16(a, b, acc);
            }
            float bsum = bip[m * H_ + c * 16 + col] + bias[m * H_ + c * 16 + col];
#pragma unroll
            for (int r = 0; r < 8; ++r) {
                float v = tanhf(acc[r] + bsum);
                sHin[rb + r][m * H_ + c * 16 + col] = f2bf_t(v);
            }
        }
        __syncthreads();

        // ---- Phase C: Q/K projections (N = 16) ----
        for (int task = wave; task < 48; task += 8) {
            int qk = task / 24;
            int am = task % 24, a = am / 6, m = am % 6;
            const unsigned short* wb =
                wbase + (qk ? OFF_WK : OFF_WQ) + ((long)a * M_ + m) * 256 * 16;
            v8f acc = zero8();
#pragma unroll
            for (int kc = 0; kc < 8; ++kc) {
                v16bf av = load_a_lds(&sHin[0][m * H_], M_ * H_, kc * 32);
                v16bf bv = load_b_pk(wb, 0, kc, 8);
                acc = wmma_bf16(av, bv, acc);
            }
            float* dst = qk ? &sKt[a][m][0][0] : &sQ[a][m][0][0];
#pragma unroll
            for (int r = 0; r < 8; ++r) dst[(rb + r) * KD_ + col] = acc[r];
        }
        __syncthreads();

        // ---- Phase D: attention weights, blend with carry ----
        for (int i = tid; i < NH_ * M_ * M_ * BT; i += 256) {
            int a   = i / (M_ * M_ * BT);
            int rem = i % (M_ * M_ * BT);
            int m   = rem / (M_ * BT);
            int n   = (rem / BT) % M_;
            int row = rem % BT;
            const v4f* qv = (const v4f*)&sQ[a][m][row][0];
            const v4f* kv = (const v4f*)&sKt[a][n][row][0];
            float d = 0.0f;
#pragma unroll
            for (int j = 0; j < 4; ++j) {
                v4f qq = qv[j], kk = kv[j];
#pragma unroll
                for (int e = 0; e < 4; ++e) d += qq[e] * kk[e];
            }
            float awv  = 1.0f / (1.0f + expf(-d * 0.25f));           // scale = 1/sqrt(16)
            float prev = sAw[a][m][n][row];
            float val  = (t == 0) ? awv : (0.5f * awv + 0.5f * prev);
            sAw[a][m][n][row] = val;
            awp[(((long)a * T_ + t) * B_ + (b0 + row)) * 36 + m * 6 + n] = val;
        }
        __syncthreads();

        // ---- Phase E: A_comm = sum_{a,n} aw * (H_in@W_V) ; H_comm = H_in + A_comm ----
        for (int c = wave; c < 16; c += 8) {
            v8f accs[M_];
#pragma unroll
            for (int m = 0; m < M_; ++m) accs[m] = zero8();
            for (int a = 0; a < NH_; ++a) {
#pragma unroll
                for (int n = 0; n < M_; ++n) {
                    v8f vc = zero8();
                    const unsigned short* wv =
                        wbase + OFF_WV + ((long)a * M_ + n) * 256 * 256;
#pragma unroll
                    for (int kc = 0; kc < 8; ++kc) {
                        v16bf av = load_a_lds(&sHin[0][n * H_], M_ * H_, kc * 32);
                        v16bf bv = load_b_pk(wv, c, kc, 8);
                        vc = wmma_bf16(av, bv, vc);
                    }
#pragma unroll
                    for (int m = 0; m < M_; ++m) {
                        v4f w0 = *(const v4f*)&sAw[a][m][n][rb];
                        v4f w1 = *(const v4f*)&sAw[a][m][n][rb + 4];
#pragma unroll
                        for (int r = 0; r < 4; ++r) {
                            accs[m][r]     += w0[r] * vc[r];
                            accs[m][r + 4] += w1[r] * vc[r + 4];
                        }
                    }
                }
            }
#pragma unroll
            for (int m = 0; m < M_; ++m)
#pragma unroll
                for (int r = 0; r < 8; ++r) {
                    int row  = rb + r;
                    int hcol = m * H_ + c * 16 + col;
                    float ac  = accs[m][r];
                    float hin = bf2f(sHin[row][hcol]);
                    float hc  = hin + ac;
                    long  idx = (((long)(b0 + row) * T_ + t) * M_ + m) * H_ + c * 16 + col;
                    acmp[idx] = ac;
                    hidp[idx] = hc;
                    sHst[row][hcol] = f2bf_t(hc);                     // next-step carry
                    if (m >= 4) sHcm[row][(m - 4) * H_ + c * 16 + col] = f2bf_t(hc);
                }
        }
        __syncthreads();

        // ---- Phase F: out = H_comm[:,4:6].reshape(.,512) @ Wout^T + bout ----
        for (int c = wave; c < 4; c += 8) {
            v8f acc = zero8();
            const unsigned short* wo = wbase + OFF_WO;
#pragma unroll
            for (int kc = 0; kc < 16; ++kc) {
                v16bf av = load_a_lds(&sHcm[0][0], 2 * H_, kc * 32);
                v16bf bv = load_b_pk(wo, c, kc, 16);
                acc = wmma_bf16(av, bv, acc);
            }
#pragma unroll
            for (int r = 0; r < 8; ++r) {
                int row = rb + r, n = c * 16 + col;
                outp[((long)(b0 + row) * T_ + t) * O_ + n] = acc[r] + bout[n];
            }
        }
        __syncthreads();
    }
}

extern "C" void kernel_launch(void* const* d_in, const int* in_sizes, int n_in,
                              void* d_out, int out_size, void* d_ws, size_t ws_size,
                              hipStream_t stream) {
    const float* x      = (const float*)d_in[0];
    const float* w_task = (const float*)d_in[1];
    const float* w_sens = (const float*)d_in[2];
    const float* wip    = (const float*)d_in[3];
    const float* bip    = (const float*)d_in[4];
    const float* u      = (const float*)d_in[5];
    const float* vlr    = (const float*)d_in[6];
    const float* bias   = (const float*)d_in[7];
    const float* w_q    = (const float*)d_in[8];
    const float* w_k    = (const float*)d_in[9];
    const float* w_v    = (const float*)d_in[10];
    const float* wout   = (const float*)d_in[11];
    const float* bout   = (const float*)d_in[12];

    unsigned short* wb = (unsigned short*)d_ws;

    struct Pk {
        const float* s; long off; int nMat; long sStride, dStride;
        int Ktot, Ntot; long strideK, strideN; int Kpad;
    };
    const Pk pk[9] = {
        {w_task, OFF_WT,   2, (long)TASK_ * H_, 32L * 256,  TASK_,    H_, H_,  1, 32 },
        {w_sens, OFF_WS,   2, (long)SENS_ * H_, 96L * 256,  SENS_,    H_, H_,  1, 96 },
        {wip,    OFF_WIP,  6, (long)H_ * H_,    65536L,     H_,       H_, H_,  1, 256},
        {u,      OFF_U,    6, (long)H_ * R_,    4096L,      H_,       R_, R_,  1, 256},
        {vlr,    OFF_VLR,  6, (long)R_ * H_,    8192L,      R_,       H_, H_,  1, 32 },
        {w_q,    OFF_WQ,  24, (long)H_ * KD_,   4096L,      H_,      KD_, KD_, 1, 256},
        {w_k,    OFF_WK,  24, (long)H_ * KD_,   4096L,      H_,      KD_, KD_, 1, 256},
        {w_v,    OFF_WV,  24, (long)H_ * H_,    65536L,     H_,       H_, H_,  1, 256},
        {wout,   OFF_WO,   1, 0L,               32768L,     2 * H_,   O_, 1, 2 * H_, 512}, // Wout^T
    };
    for (int i = 0; i < 9; ++i) {
        long total = (long)pk[i].nMat * (pk[i].Ntot / 16) * (pk[i].Kpad / 32) * 512;
        rim_pack_b_kernel<<<(unsigned)((total + 255) / 256), 256, 0, stream>>>(
            pk[i].s, wb + pk[i].off, pk[i].nMat, pk[i].sStride, pk[i].dStride,
            pk[i].Ktot, pk[i].Ntot, pk[i].strideK, pk[i].strideN, pk[i].Kpad);
    }

    float* outp = (float*)d_out;                       // (B,T,O)
    float* hidp = outp + (long)B_ * T_ * O_;           // (B,T,M,H)
    float* ainp = hidp + (long)B_ * T_ * M_ * H_;      // (B,T,M,H)
    float* acmp = ainp + (long)B_ * T_ * M_ * H_;      // (B,T,M,H)
    float* awp  = acmp + (long)B_ * T_ * M_ * H_;      // (NH,T,B,M,M)

    rim_scan_kernel<<<dim3(B_ / BT), dim3(256), 0, stream>>>(
        x, bip, bias, bout, wb, outp, hidp, ainp, acmp, awp);
}
